// CoAttention_TextDNS_13391708029220
// MI455X (gfx1250) — matrix operand order repaired and checked by reading, not compiled, not run on hardware
//
#include <hip/hip_runtime.h>
#include <math.h>

typedef __attribute__((ext_vector_type(16))) _Float16 v16h;
typedef __attribute__((ext_vector_type(8)))  _Float16 v8h;
typedef __attribute__((ext_vector_type(4)))  _Float16 v4h;
typedef __attribute__((ext_vector_type(8)))  float    v8f;

#define B_ 8
#define L_ 256
#define M_ 128
#define H_ 768
#define WN_ (H_ * H_)                   // 589824 elements per weight matrix

// Workspace layout:
//   floats [0, 16384): scores + attention vectors (64 KB)
//   bytes  [65536, 65536 + 4*WN*2): packed hi/lo f16 weights (optional, 4.5 MB)
#define WS_SC_TEXT 0                    // B*L  = 2048 scores
#define WS_SC_DNS  (B_ * L_)            // B*M  = 1024 scores
#define WS_VEC_TEXT 4096                // B*H  = 6144 floats
#define WS_VEC_DNS  (4096 + B_ * H_)    // B*H  = 6144 floats
#define WS_WPACK_BYTES 65536

// Fast tanh: gfx1250 V_TANH_F32 (TRANS32, co-executes with WMMA pipe).
__device__ __forceinline__ float fast_tanhf(float x) {
#if __has_builtin(__builtin_amdgcn_tanhf)
    return __builtin_amdgcn_tanhf(x);
#elif __has_builtin(__builtin_amdgcn_tanh_f32)
    return __builtin_amdgcn_tanh_f32(x);
#else
    const float ax = fabsf(x);
    const float e  = __expf(2.0f * ax);
    const float t  = 1.0f - 2.0f / (e + 1.0f);
    return copysignf(t, x);
#endif
}

// ---------------------------------------------------------------------------
// Kernel 0 (optional): pre-split both weight matrices into hi/lo f16 once.
// wpack layout: [Wt2_hi | Wt2_lo | Wd1_hi | Wd1_lo], each WN_ halfs.
// ---------------------------------------------------------------------------
__global__ __launch_bounds__(256) void coatt_wpack_kernel(
    const float* __restrict__ Wt2, const float* __restrict__ Wd1,
    _Float16* __restrict__ wpack)
{
    const int i4  = blockIdx.x * 256 + threadIdx.x;   // 294912 float4's total
    const int e   = i4 * 4;
    const int m   = (e >= WN_) ? 1 : 0;
    const int off = e - m * WN_;
    const float4 v = *(const float4*)((m ? Wd1 : Wt2) + off);
    const float f[4] = {v.x, v.y, v.z, v.w};
    v4h hi, lo;
#pragma unroll
    for (int k = 0; k < 4; ++k) {
        const _Float16 h = (_Float16)f[k];
        hi[k] = h;
        lo[k] = (_Float16)(f[k] - (float)h);
    }
    _Float16* dhi = wpack + (size_t)m * 2 * WN_ + off;
    *(v4h*)dhi        = hi;
    *(v4h*)(dhi + WN_) = lo;
}

// ---------------------------------------------------------------------------
// Kernel 1: fused GEMM + tanh + weighted column-reduction producing scores
//   s[row] = sum_o tanh( feat[row,:] . W[o,:] ) * wv[o]
// One block = 16 rows x 768 o-cols, K = 768.  8 waves, each owns 96 o-cols.
// f32 emulated via f16 hi/lo split: A.B ~= Ahi.Bhi + Ahi.Blo + Alo.Bhi.
// PACKED=true reads pre-split f16 weights (no per-iteration conversion VALU,
// same L2 weight bytes as f32 reads).
// ---------------------------------------------------------------------------
template <bool PACKED>
__global__ __launch_bounds__(256) void coatt_score_kernel(
    const float* __restrict__ text, const float* __restrict__ dns,
    const float* __restrict__ Wt2,  const float* __restrict__ Wd1,
    const _Float16* __restrict__ wpack,
    const float* __restrict__ watt2, const float* __restrict__ watt1,
    float* __restrict__ ws)
{
    __shared__ __align__(16) _Float16 Ahi[16 * H_];   // 24 KB
    __shared__ __align__(16) _Float16 Alo[16 * H_];   // 24 KB
    __shared__ float s_tile[16];

    const int bx = blockIdx.x;
    int b, tile, R;
    const float* feat; const float* W; const float* wv; float* sout;
    const _Float16* whi; const _Float16* wlo;
    if (bx < 128) {                       // task 0: text -> t2 scores (wd)
        b = bx >> 4; tile = bx & 15; R = L_;
        feat = text; W = Wt2; wv = watt2 + H_;
        whi = wpack;               wlo = wpack + WN_;
        sout = ws + WS_SC_TEXT + b * L_;
    } else {                              // task 1: dns -> d1 scores (wb)
        const int t = bx - 128;
        b = t >> 3; tile = t & 7; R = M_;
        feat = dns; W = Wd1; wv = watt1 + H_;
        whi = wpack + 2 * WN_;     wlo = wpack + 3 * WN_;
        sout = ws + WS_SC_DNS + b * M_;
    }
    const float* Arows = feat + ((size_t)b * R + (size_t)tile * 16) * H_;

    // Stage 16x768 A tile into LDS as f16 hi/lo split (exact to ~2^-22).
    for (int i = threadIdx.x; i < 16 * H_ / 4; i += 256) {
        const float4 v = ((const float4*)Arows)[i];
        const float f[4] = {v.x, v.y, v.z, v.w};
        v4h hi, lo;
#pragma unroll
        for (int e = 0; e < 4; ++e) {
            const _Float16 h = (_Float16)f[e];
            hi[e] = h;
            lo[e] = (_Float16)(f[e] - (float)h);
        }
        *(v4h*)&Ahi[i * 4] = hi;
        *(v4h*)&Alo[i * 4] = lo;
    }
    if (threadIdx.x < 16) s_tile[threadIdx.x] = 0.0f;
    __syncthreads();

    const int wave = threadIdx.x >> 5;
    const int lane = threadIdx.x & 31;
    const int kh   = lane >> 4;      // lane-half selects K sub-block
    const int nm   = lane & 15;      // A: row M ; B/C: col N

    for (int ot = 0; ot < 6; ++ot) {
        const int obase = wave * 96 + ot * 16;
        const int o     = obase + nm;
        const float*    wrow = W   + (size_t)o * H_ + kh * 16;  // B col n = W row o
        const _Float16* hrow = whi + (size_t)o * H_ + kh * 16;
        const _Float16* lrow = wlo + (size_t)o * H_ + kh * 16;

        v8f c = {};
        for (int k0 = 0; k0 < H_; k0 += 32) {
            // ---- B fragment: lane(n) elems i = B[k0 + kh*16 + i][n] = W[o][k]
            v16h bhi, blo;
            if constexpr (PACKED) {
                bhi = __builtin_shufflevector(*(const v8h*)&hrow[k0],
                                              *(const v8h*)&hrow[k0 + 8],
                    0,1,2,3,4,5,6,7,8,9,10,11,12,13,14,15);
                blo = __builtin_shufflevector(*(const v8h*)&lrow[k0],
                                              *(const v8h*)&lrow[k0 + 8],
                    0,1,2,3,4,5,6,7,8,9,10,11,12,13,14,15);
            } else {
#pragma unroll
                for (int e = 0; e < 16; ++e) {
                    const float w = wrow[k0 + e];
                    const _Float16 h = (_Float16)w;
                    bhi[e] = h;
                    blo[e] = (_Float16)(w - (float)h);
                }
            }
            // ---- A fragment: row m = nm; elems i = A[nm][k0 + kh*16 + i]
            const int ka = k0 + kh * 16;
            const v8h ahl = *(const v8h*)&Ahi[nm * H_ + ka];
            const v8h ahh = *(const v8h*)&Ahi[nm * H_ + ka + 8];
            const v8h all = *(const v8h*)&Alo[nm * H_ + ka];
            const v8h alh = *(const v8h*)&Alo[nm * H_ + ka + 8];
            const v16h ahi = __builtin_shufflevector(ahl, ahh,
                0,1,2,3,4,5,6,7,8,9,10,11,12,13,14,15);
            const v16h alo = __builtin_shufflevector(all, alh,
                0,1,2,3,4,5,6,7,8,9,10,11,12,13,14,15);

            c = __builtin_amdgcn_wmma_f32_16x16x32_f16(false, ahi, false, bhi,
                                                       (short)0, c, false, false);
            c = __builtin_amdgcn_wmma_f32_16x16x32_f16(false, ahi, false, blo,
                                                       (short)0, c, false, false);
            c = __builtin_amdgcn_wmma_f32_16x16x32_f16(false, alo, false, bhi,
                                                       (short)0, c, false, false);
        }

        // Epilogue: s[m] += sum_n tanh(C[m,n]) * wv[obase + n]
        // C layout: lane n = nm holds rows m = kh*8 + r in c[r].
        const float wo = wv[o];
        float acc[8];
#pragma unroll
        for (int r = 0; r < 8; ++r) acc[r] = fast_tanhf(c[r]) * wo;
#pragma unroll
        for (int mask = 1; mask < 16; mask <<= 1) {
#pragma unroll
            for (int r = 0; r < 8; ++r) acc[r] += __shfl_xor(acc[r], mask, 32);
        }
        if (nm == 0) {
            const int rb = kh * 8;
#pragma unroll
            for (int r = 0; r < 8; ++r) atomicAdd(&s_tile[rb + r], acc[r]);
        }
    }
    __syncthreads();
    if (threadIdx.x < 16) sout[tile * 16 + threadIdx.x] = s_tile[threadIdx.x];
}

// ---------------------------------------------------------------------------
// Kernel 2: per (batch, task) softmax over rows + weighted feature sum.
//   vec[b,h] = sum_r softmax(s)[r] * feat[b,r,h]
// ---------------------------------------------------------------------------
__global__ __launch_bounds__(256) void coatt_reduce_kernel(
    const float* __restrict__ text, const float* __restrict__ dns,
    float* __restrict__ ws)
{
    __shared__ float red[256];
    __shared__ float aw[256];
    const int bx = blockIdx.x;
    const int t  = threadIdx.x;
    int b, R; const float* feat; const float* s; float* vec;
    if (bx < 8) { b = bx;     R = L_; feat = text; s = ws + WS_SC_TEXT + b * L_; vec = ws + WS_VEC_TEXT + b * H_; }
    else        { b = bx - 8; R = M_; feat = dns;  s = ws + WS_SC_DNS  + b * M_; vec = ws + WS_VEC_DNS  + b * H_; }

    const float sv = (t < R) ? s[t] : -1e30f;
    red[t] = sv; __syncthreads();
    for (int off = 128; off > 0; off >>= 1) {
        if (t < off) red[t] = fmaxf(red[t], red[t + off]);
        __syncthreads();
    }
    const float mx = red[0]; __syncthreads();
    const float e = (t < R) ? __expf(sv - mx) : 0.0f;
    aw[t] = e; red[t] = e; __syncthreads();
    for (int off = 128; off > 0; off >>= 1) {
        if (t < off) red[t] += red[t + off];
        __syncthreads();
    }
    const float inv = 1.0f / red[0];

    const float* F = feat + (size_t)b * R * H_;
    for (int h = t; h < H_; h += 256) {
        float acc = 0.0f;
        for (int r = 0; r < R; ++r) acc += aw[r] * F[(size_t)r * H_ + h];
        vec[h] = acc * inv;
    }
}

// ---------------------------------------------------------------------------
// Kernel 3: broadcast per-batch vectors across L into both outputs
// (att_text then att_dns, each (B,L,H)) with float4 stores.
// ---------------------------------------------------------------------------
__global__ __launch_bounds__(256) void coatt_broadcast_kernel(
    const float* __restrict__ ws, float* __restrict__ out)
{
    const int i4 = blockIdx.x * 256 + threadIdx.x;        // 786432 float4's
    const size_t e   = (size_t)i4 * 4;
    const size_t BLH = (size_t)B_ * L_ * H_;
    const int task   = (e >= BLH) ? 1 : 0;                // 0 = att_text
    const size_t rem = e - (size_t)task * BLH;
    const int b = (int)(rem / (L_ * H_));
    const int h = (int)(rem % H_);
    const float* vec = ws + (task ? WS_VEC_DNS : WS_VEC_TEXT) + b * H_;
    ((float4*)out)[i4] = *(const float4*)&vec[h];
}

// ---------------------------------------------------------------------------
extern "C" void kernel_launch(void* const* d_in, const int* in_sizes, int n_in,
                              void* d_out, int out_size, void* d_ws, size_t ws_size,
                              hipStream_t stream)
{
    (void)in_sizes; (void)n_in; (void)out_size;
    const float* text  = (const float*)d_in[0];
    const float* dns   = (const float*)d_in[1];
    const float* Wd1   = (const float*)d_in[4];
    const float* watt1 = (const float*)d_in[5];
    const float* Wt2   = (const float*)d_in[9];
    const float* watt2 = (const float*)d_in[10];
    float* ws  = (float*)d_ws;
    float* out = (float*)d_out;

    const size_t need = WS_WPACK_BYTES + (size_t)4 * WN_ * sizeof(_Float16);
    if (ws_size >= need) {
        _Float16* wpack = (_Float16*)((char*)d_ws + WS_WPACK_BYTES);
        coatt_wpack_kernel<<<2 * WN_ / (4 * 256), 256, 0, stream>>>(Wt2, Wd1, wpack);
        coatt_score_kernel<true><<<192, 256, 0, stream>>>(
            text, dns, Wt2, Wd1, wpack, watt2, watt1, ws);
    } else {
        coatt_score_kernel<false><<<192, 256, 0, stream>>>(
            text, dns, Wt2, Wd1, (const _Float16*)nullptr, watt2, watt1, ws);
    }
    coatt_reduce_kernel<<<16, 256, 0, stream>>>(text, dns, ws);
    coatt_broadcast_kernel<<<(2 * B_ * L_ * H_) / (4 * 256), 256, 0, stream>>>(ws, out);
}
